// FeatureAlign_85212151153053
// MI455X (gfx1250) — matrix operand order repaired and checked by reading, not compiled, not run on hardware
//
#include <hip/hip_runtime.h>
#include <hip/hip_bf16.h>

// ---------------------------------------------------------------------------
// FeatureAlign: offset(1x1 conv) -> deformable 3x3 conv (im2col + bf16 WMMA
// GEMM) -> GroupNorm(32) + ReLU.   B=4, C=256, H=W=64, DG=4, K=3.
// GEMM per batch: out[256,4096] = W[256,2304] x S[2304,4096], Kflat =
// dg*576 + c*9 + k  (== w_deform's natural memory order).
// Round 2: hoist all bilinear geometry (masked weights + clamped indices)
// into the per-block precompute phase; staging loop is now pure gather+FMA.
// ---------------------------------------------------------------------------

typedef __attribute__((ext_vector_type(16))) __bf16        v16bf;
typedef __attribute__((ext_vector_type(8)))  float         v8f;
typedef __attribute__((ext_vector_type(4)))  unsigned int  v4u;
typedef __attribute__((ext_vector_type(8)))  unsigned int  v8u;

#define HW      4096
#define KTOT    2304
#define KCHUNKS 72      // 2304 / 32

__device__ __forceinline__ unsigned short f2bf(float f) {
    unsigned int u = __float_as_uint(f);
    u += 0x7fffu + ((u >> 16) & 1u);           // round to nearest even
    return (unsigned short)(u >> 16);
}

__device__ __forceinline__ v16bf load_frag(const unsigned short* p) {
    const v4u* q = (const v4u*)p;              // 16B-aligned by construction
    v4u lo = q[0], hi = q[1];
    v8u r = {lo.x, lo.y, lo.z, lo.w, hi.x, hi.y, hi.z, hi.w};
    return __builtin_bit_cast(v16bf, r);
}

// ---------------------------------------------------------------------------
// Kernel 1: pack w_deform [256][2304] f32 into bf16 A-fragments.
// apack[((kc*16 + mt)*32 + lane)*16 + i] uses contiguous k order (matching the B fragments):
//   m = mt*16 + (lane&15); half = lane>>4;
//   K = kc*32 + half*16 + i
// ---------------------------------------------------------------------------
__global__ __launch_bounds__(256) void pack_w_kernel(
    const float* __restrict__ w, unsigned short* __restrict__ apack) {
    int t    = blockIdx.x * 256 + threadIdx.x;   // 0..36863
    int lane = t & 31;
    int mt   = (t >> 5) & 15;
    int kc   = t >> 9;
    int m    = mt * 16 + (lane & 15);
    int half = lane >> 4;
    const float* src = w + (size_t)m * KTOT + kc * 32 + half * 16;
    unsigned short* dst = apack + (size_t)t * 16;
#pragma unroll
    for (int i = 0; i < 8; ++i) dst[i]     = f2bf(src[i]);
#pragma unroll
    for (int i = 0; i < 8; ++i) dst[8 + i] = f2bf(src[8 + i]);
}

// ---------------------------------------------------------------------------
// Kernel 2: fused offset + bilinear im2col + WMMA GEMM.
// grid = B * (HW/64) = 256 blocks, 256 threads (8 waves).
// Block computes y[b, 0:256, p0:p0+64].
// ---------------------------------------------------------------------------
__global__ __launch_bounds__(256) void deform_gemm_kernel(
    const float* __restrict__ x, const float* __restrict__ shape,
    const float* __restrict__ w_off, const unsigned short* __restrict__ apack,
    float* __restrict__ y) {
    __shared__ float          s_shape[4 * 64];   // shape staged per pixel
    __shared__ float          s_woff[288];       // 72x4 1x1-conv weights
    __shared__ float4         s_wts[2304];       // 4 masked bilinear weights
    __shared__ uint2          s_idx[2304];       // 4 clamped idx (4 x u16)
    __shared__ unsigned short s_B[64 * 40];      // B tile [n=64][k=32], pad 40

    const int tid = threadIdx.x;
    const int b   = blockIdx.x >> 6;
    const int p0  = (blockIdx.x & 63) * 64;

    // ---- stage shape (4x64 pixels) and w_offset (288 floats) ----
    {
        int i = tid >> 6, p = tid & 63;
        s_shape[i * 64 + p] = shape[(size_t)(b * 4 + i) * HW + p0 + p];
    }
    for (int t = tid; t < 288; t += 256) s_woff[t] = w_off[t];
    __syncthreads();

    // ---- precompute bilinear geometry per (dg,k,pixel): 2304 entries ----
    for (int e = tid; e < 2304; e += 256) {
        int p   = e & 63;
        int dk  = e >> 6;                 // dg*9 + k
        int k   = dk % 9;
        int pix = p0 + p;
        int h   = pix >> 6, w = pix & 63;
        float s0 = s_shape[p], s1 = s_shape[64 + p];
        float s2 = s_shape[128 + p], s3 = s_shape[192 + p];
        const float* wr = s_woff + dk * 8;     // dy row then dx row (4 each)
        float offy = wr[0] * s0 + wr[1] * s1 + wr[2] * s2 + wr[3] * s3;
        float offx = wr[4] * s0 + wr[5] * s1 + wr[6] * s2 + wr[7] * s3;
        float py = (float)(h + k / 3 - 1) + offy;
        float px = (float)(w + k % 3 - 1) + offx;
        float yf = floorf(py), xf = floorf(px);
        float wy1 = py - yf, wx1 = px - xf;
        float wy0 = 1.f - wy1, wx0 = 1.f - wx1;
        int y0 = (int)yf, x0 = (int)xf;
        int y1 = y0 + 1, x1 = x0 + 1;
        float vy0 = (y0 >= 0 && y0 < 64) ? 1.f : 0.f;
        float vy1 = (y1 >= 0 && y1 < 64) ? 1.f : 0.f;
        float vx0 = (x0 >= 0 && x0 < 64) ? 1.f : 0.f;
        float vx1 = (x1 >= 0 && x1 < 64) ? 1.f : 0.f;
        int iy0 = min(max(y0, 0), 63), iy1 = min(max(y1, 0), 63);
        int ix0 = min(max(x0, 0), 63), ix1 = min(max(x1, 0), 63);
        float4 wt;
        wt.x = wy0 * wx0 * vy0 * vx0;
        wt.y = wy0 * wx1 * vy0 * vx1;
        wt.z = wy1 * wx0 * vy1 * vx0;
        wt.w = wy1 * wx1 * vy1 * vx1;
        s_wts[e] = wt;
        unsigned i00 = (unsigned)(iy0 * 64 + ix0);
        unsigned i01 = (unsigned)(iy0 * 64 + ix1);
        unsigned i10 = (unsigned)(iy1 * 64 + ix0);
        unsigned i11 = (unsigned)(iy1 * 64 + ix1);
        s_idx[e] = make_uint2(i00 | (i01 << 16), i10 | (i11 << 16));
    }
    __syncthreads();

    const int wave = tid >> 5, lane = tid & 31;
    const int p_fx = tid & 63;       // pixel handled in staging (loop-invariant)
    const int kk_b = tid >> 6;       // kk = e8*4 + kk_b

    v8f acc[2][4];
#pragma unroll
    for (int mi = 0; mi < 2; ++mi)
#pragma unroll
        for (int nt = 0; nt < 4; ++nt) {
            v8f z = {0.f, 0.f, 0.f, 0.f, 0.f, 0.f, 0.f, 0.f};
            acc[mi][nt] = z;
        }

    const float* xb_base = x + (size_t)b * 256 * HW;

    for (int kc = 0; kc < KCHUNKS; ++kc) {
        __syncthreads();                       // s_B consumed by prev iter
        // ---- stage B tile: 32(K) x 64(N) bilinear samples -> bf16 LDS ----
#pragma unroll
        for (int e8 = 0; e8 < 8; ++e8) {
            int kk   = e8 * 4 + kk_b;
            int kidx = kc * 32 + kk;           // dg*576 + c*9 + k
            int dg   = kidx / 576;
            int r    = kidx - dg * 576;
            int c    = r / 9;
            int k    = r - c * 9;
            int e    = (dg * 9 + k) * 64 + p_fx;
            float4 wt = s_wts[e];
            uint2  id = s_idx[e];
            const float* xb = xb_base + (size_t)((dg << 6) + c) * HW;
            float g = wt.x * xb[id.x & 0xffffu]
                    + wt.y * xb[id.x >> 16]
                    + wt.z * xb[id.y & 0xffffu]
                    + wt.w * xb[id.y >> 16];
            s_B[p_fx * 40 + kk] = f2bf(g);
        }
        __syncthreads();

        // ---- B fragments from LDS (ds_load_b128 x2 per frag) ----
        v16bf bfrag[4];
#pragma unroll
        for (int nt = 0; nt < 4; ++nt) {
            int row = nt * 16 + (lane & 15);
            bfrag[nt] = load_frag(&s_B[row * 40 + (lane >> 4) * 16]);
        }
        // ---- A fragments from prepacked global + 8 WMMAs per wave ----
#pragma unroll
        for (int mi = 0; mi < 2; ++mi) {
            int mt = wave * 2 + mi;
            v16bf afrag =
                load_frag(apack + ((size_t)(kc * 16 + mt) * 32 + lane) * 16);
#pragma unroll
            for (int nt = 0; nt < 4; ++nt) {
                acc[mi][nt] = __builtin_amdgcn_wmma_f32_16x16x32_bf16(
                    false, afrag, false, bfrag[nt], (short)0, acc[mi][nt],
                    false, false);
            }
        }
    }

    // ---- writeback: C/D layout lane l -> n=l%16, rows r + 8*(l>>4) ----
#pragma unroll
    for (int mi = 0; mi < 2; ++mi) {
        int mbase = wave * 32 + mi * 16 + ((lane >> 4) << 3);
#pragma unroll
        for (int nt = 0; nt < 4; ++nt) {
            int p = p0 + nt * 16 + (lane & 15);
#pragma unroll
            for (int r2 = 0; r2 < 8; ++r2) {
                y[(size_t)(b * 256 + mbase + r2) * HW + p] = acc[mi][nt][r2];
            }
        }
    }
}

// ---------------------------------------------------------------------------
// Kernel 3: GroupNorm(32) + affine + ReLU.  grid = B*32 blocks.
// ---------------------------------------------------------------------------
__global__ __launch_bounds__(256) void gn_relu_kernel(
    const float* __restrict__ y, const float* __restrict__ gamma,
    const float* __restrict__ beta, float* __restrict__ out) {
    constexpr int N = 8 * HW;                  // elems per (b, group)
    int b = blockIdx.x >> 5;
    int g = blockIdx.x & 31;
    const float* base = y + (size_t)(b * 256 + g * 8) * HW;
    float sum = 0.f, ss = 0.f;
    for (int t = threadIdx.x; t < N; t += 256) {
        float v = base[t];
        sum += v;
        ss += v * v;
    }
    __shared__ float s_sum[256], s_ss[256];
    s_sum[threadIdx.x] = sum;
    s_ss[threadIdx.x]  = ss;
    __syncthreads();
    for (int s = 128; s > 0; s >>= 1) {
        if (threadIdx.x < s) {
            s_sum[threadIdx.x] += s_sum[threadIdx.x + s];
            s_ss[threadIdx.x]  += s_ss[threadIdx.x + s];
        }
        __syncthreads();
    }
    float mu  = s_sum[0] * (1.f / N);
    float var = s_ss[0] * (1.f / N) - mu * mu;
    float rs  = rsqrtf(var + 1e-5f);
    float* ob = out + (size_t)(b * 256 + g * 8) * HW;
    for (int t = threadIdx.x; t < N; t += 256) {
        int c = g * 8 + (t >> 12);
        float v = (base[t] - mu) * rs * gamma[c] + beta[c];
        ob[t] = v > 0.f ? v : 0.f;
    }
}

extern "C" void kernel_launch(void* const* d_in, const int* in_sizes, int n_in,
                              void* d_out, int out_size, void* d_ws,
                              size_t ws_size, hipStream_t stream) {
    const float* x        = (const float*)d_in[0];  // [4,256,64,64]
    const float* shape    = (const float*)d_in[1];  // [4,4,64,64]
    const float* w_offset = (const float*)d_in[2];  // [72,4,1,1]
    const float* w_deform = (const float*)d_in[3];  // [256,256,3,3]
    const float* gamma    = (const float*)d_in[4];  // [256]
    const float* beta     = (const float*)d_in[5];  // [256]
    float* out = (float*)d_out;

    char* ws = (char*)d_ws;
    float* ws_y = (float*)ws;                                   // 16.78 MB
    unsigned short* apack =
        (unsigned short*)(ws + (size_t)4 * 256 * HW * sizeof(float));  // 1.18 MB

    pack_w_kernel<<<144, 256, 0, stream>>>(w_deform, apack);
    deform_gemm_kernel<<<256, 256, 0, stream>>>(x, shape, w_offset, apack,
                                                ws_y);
    gn_relu_kernel<<<128, 256, 0, stream>>>(ws_y, gamma, beta, out);
}
